// VQVAE_36644660969914
// MI455X (gfx1250) — matrix operand order repaired and checked by reading, not compiled, not run on hardware
//
#include <hip/hip_runtime.h>
#include <hip/hip_bf16.h>

#define BATCH      32768
#define INPUT_DIM  1024
#define HIDDEN_DIM 2048
#define LATENT_DIM 64
#define NUM_EMB    4096

typedef __attribute__((ext_vector_type(16))) __bf16 v16bf;
typedef __attribute__((ext_vector_type(8)))  __bf16 v8bf;
typedef __attribute__((ext_vector_type(8)))  float  v8f;

__device__ __forceinline__ unsigned short f32_to_bf16_raw(float f) {
    unsigned u = __builtin_bit_cast(unsigned, f);
    u += 0x7FFFu + ((u >> 16) & 1u);         // round-to-nearest-even
    return (unsigned short)(u >> 16);
}

// Async 16B copy global -> LDS (ASYNCcnt, no VGPR round trip).  Generic
// pointers to __shared__ carry the LDS offset in their low 32 bits.
__device__ __forceinline__ void async_copy_b128(void* lds_dst, const void* gsrc) {
    const unsigned lds_off = (unsigned)(size_t)lds_dst;
    const unsigned long long ga = (unsigned long long)(size_t)gsrc;
    asm volatile("global_load_async_to_lds_b128 %0, %1, off"
                 :: "v"(lds_off), "v"(ga) : "memory");
}

// ---------------------------------------------------------------------------
// Elementwise f32 -> raw bf16, 8 elements/thread.
// ---------------------------------------------------------------------------
__global__ __launch_bounds__(256) void f32_to_bf16_kernel(
    const float* __restrict__ in, unsigned short* __restrict__ out)
{
    const size_t i = ((size_t)blockIdx.x * 256 + threadIdx.x) * 8;
    const float4 a = *(const float4*)(in + i);
    const float4 b = *(const float4*)(in + i + 4);
    uint4 o;
    o.x = (unsigned)f32_to_bf16_raw(a.x) | ((unsigned)f32_to_bf16_raw(a.y) << 16);
    o.y = (unsigned)f32_to_bf16_raw(a.z) | ((unsigned)f32_to_bf16_raw(a.w) << 16);
    o.z = (unsigned)f32_to_bf16_raw(b.x) | ((unsigned)f32_to_bf16_raw(b.y) << 16);
    o.w = (unsigned)f32_to_bf16_raw(b.z) | ((unsigned)f32_to_bf16_raw(b.w) << 16);
    *(uint4*)(out + i) = o;
}

// ---------------------------------------------------------------------------
// Transpose + convert W[K,N] (f32) -> Wt[N,K] (raw bf16), LDS-tiled.
// ---------------------------------------------------------------------------
__global__ __launch_bounds__(256) void transpose_to_bf16(
    const float* __restrict__ W, unsigned short* __restrict__ Wt, int K, int N)
{
    __shared__ float tile[32][33];
    const int n0 = blockIdx.x * 32, k0 = blockIdx.y * 32;
    const int tx = threadIdx.x & 31, ty = threadIdx.x >> 5;
#pragma unroll
    for (int i = 0; i < 32; i += 8)
        tile[ty + i][tx] = W[(size_t)(k0 + ty + i) * N + n0 + tx];
    __syncthreads();
#pragma unroll
    for (int i = 0; i < 32; i += 8)
        Wt[(size_t)(n0 + ty + i) * K + k0 + tx] = f32_to_bf16_raw(tile[tx][ty + i]);
}

// ---------------------------------------------------------------------------
// Codebook row norms: esq[k] = sum_i cb[k][i]^2.
// ---------------------------------------------------------------------------
__global__ __launch_bounds__(256) void codebook_esq_kernel(
    const float* __restrict__ cb, float* __restrict__ esq)
{
    const int k = blockIdx.x * 256 + threadIdx.x;
    float s = 0.f;
#pragma unroll
    for (int i = 0; i < LATENT_DIM; i += 4) {
        const float4 v = *(const float4*)(cb + (size_t)k * LATENT_DIM + i);
        s = fmaf(v.x, v.x, fmaf(v.y, v.y, fmaf(v.z, v.z, fmaf(v.w, v.w, s))));
    }
    esq[k] = s;
}

// ---------------------------------------------------------------------------
// GEMM: C[M,N] = act(A[M,K] @ B[K,N] + bias); A bf16 [M,K], Bt bf16 [N,K].
// 8 waves, wave tile (WTM*16)x(WTN*16), K step 32, double-buffered LDS fed
// by global_load_async_to_lds_b128.
// ---------------------------------------------------------------------------
template<int BM, int BN, int WTM, int WTN, bool RELU, bool OUT_BF16>
__global__ __launch_bounds__(256, 2) void gemm_wmma(
    const unsigned short* __restrict__ A, const unsigned short* __restrict__ Bt,
    const float* __restrict__ bias, void* __restrict__ C_,
    int M, int N, int K)
{
    constexpr int WAVES_M = BM / (WTM * 16);
    constexpr int WAVES_N = BN / (WTN * 16);
    static_assert(WAVES_M * WAVES_N == 8, "need exactly 8 waves");
    static_assert((BM * 4) % 256 == 0 && (BN * 4) % 256 == 0, "chunk counts");
    constexpr int LDT = 40;
    __shared__ __align__(16) unsigned short As[2][BM][LDT];
    __shared__ __align__(16) unsigned short Bs[2][BN][LDT];

    const int tid    = threadIdx.x;
    const int blockM = blockIdx.y * BM;
    const int blockN = blockIdx.x * BN;
    const int wave   = tid >> 5;
    const int lane   = tid & 31;
    const int rowBase = (wave % WAVES_M) * (WTM * 16);
    const int colBase = (wave / WAVES_M) * (WTN * 16);

    const int mA  = lane & 15;
    const int kHi = (lane >> 4) * 8;
    const int nB  = lane & 15;
    const int kB  = (lane >> 4) * 16;

    constexpr int ACH = (BM * 4) / 256;
    constexpr int BCH = (BN * 4) / 256;
    constexpr int NCH = ACH + BCH;

    auto stage = [&](int kk, int buf) {
#pragma unroll
        for (int it = 0; it < ACH; ++it) {
            const int c = it * 256 + tid;
            const int r = c >> 2, kc = c & 3;
            async_copy_b128(&As[buf][r][kc * 8],
                            A + (size_t)(blockM + r) * K + kk + kc * 8);
        }
#pragma unroll
        for (int it = 0; it < BCH; ++it) {
            const int c = it * 256 + tid;
            const int n = c >> 2, kc = c & 3;
            async_copy_b128(&Bs[buf][n][kc * 8],
                            Bt + (size_t)(blockN + n) * K + kk + kc * 8);
        }
    };

    v8f acc[WTM][WTN] = {};

    stage(0, 0);
    const int T = K / 32;
    for (int t = 0; t < T; ++t) {
        const int cur = t & 1;
        if (t + 1 < T) {
            stage((t + 1) * 32, cur ^ 1);
            asm volatile("s_wait_asynccnt %0" :: "n"(NCH) : "memory");
        } else {
            asm volatile("s_wait_asynccnt 0" ::: "memory");
        }
        __syncthreads();

        v16bf a[WTM], b[WTN];
#pragma unroll
        for (int i = 0; i < WTM; ++i) {
            v8bf lo = *(const v8bf*)&As[cur][rowBase + i * 16 + mA][kHi];
            v8bf hi = *(const v8bf*)&As[cur][rowBase + i * 16 + mA][16 + kHi];
            a[i] = __builtin_shufflevector(lo, hi, 0,1,2,3,4,5,6,7,8,9,10,11,12,13,14,15);
        }
#pragma unroll
        for (int j = 0; j < WTN; ++j) {
            v8bf lo = *(const v8bf*)&Bs[cur][colBase + j * 16 + nB][kHi];
            v8bf hi = *(const v8bf*)&Bs[cur][colBase + j * 16 + nB][16 + kHi];
            b[j] = __builtin_shufflevector(lo, hi, 0,1,2,3,4,5,6,7,8,9,10,11,12,13,14,15);
        }
#pragma unroll
        for (int i = 0; i < WTM; ++i)
#pragma unroll
            for (int j = 0; j < WTN; ++j)
                acc[i][j] = __builtin_amdgcn_wmma_f32_16x16x32_bf16(
                    false, a[i], false, b[j], (short)0, acc[i][j], false, false);

        __syncthreads();
    }

    const int mOff = (lane >> 4) * 8;
#pragma unroll
    for (int j = 0; j < WTN; ++j) {
        const int col = blockN + colBase + j * 16 + nB;
        const float bv = bias[col];
#pragma unroll
        for (int i = 0; i < WTM; ++i) {
#pragma unroll
            for (int r = 0; r < 8; ++r) {
                const int row = blockM + rowBase + i * 16 + mOff + r;
                float v = acc[i][j][r] + bv;
                if (RELU) v = fmaxf(v, 0.f);
                if (OUT_BF16)
                    ((unsigned short*)C_)[(size_t)row * N + col] = f32_to_bf16_raw(v);
                else
                    ((float*)C_)[(size_t)row * N + col] = v;
            }
        }
    }
}

// ---------------------------------------------------------------------------
// VQ scoring via WMMA:  argmin_k ( esq[k] - 2 * z_e . cb[k] ).
// Block = 64 rows x 4096 codes.  z rows staged to LDS once (fragments then
// hoisted to registers); codebook streamed in 256-code tiles through
// double-buffered async-to-LDS.  Per-row argmin kept in registers, reduced
// across the 16 lanes of each half-wave (D layout: one row's 16 code
// columns live in one half; halves hold disjoint rows).
// ---------------------------------------------------------------------------
__global__ __launch_bounds__(256, 2) void vq_score_argmin(
    const unsigned short* __restrict__ zbf,   // [B,64] bf16
    const unsigned short* __restrict__ cbbf,  // [4096,64] bf16
    const float* __restrict__ esq,            // [4096]
    int* __restrict__ besti_out)              // [B]
{
    constexpr int LDT = 72;                         // 64 + pad, 144B rows
    constexpr int CT  = 256;                        // codes per tile
    constexpr int NT  = NUM_EMB / CT;               // 16 tiles
    __shared__ __align__(16) unsigned short As[64][LDT];
    __shared__ __align__(16) unsigned short Bs[2][CT][LDT];

    const int tid    = threadIdx.x;
    const int blockM = blockIdx.x * 64;
    const int wave   = tid >> 5;
    const int lane   = tid & 31;
    const int rowBase = (wave & 1) * 32;            // WAVES_M = 2
    const int colBase = (wave >> 1) * 64;           // WAVES_N = 4, WTN = 4

    const int mA  = lane & 15;
    const int kHi = (lane >> 4) * 8;
    const int nB  = lane & 15;
    const int kB  = (lane >> 4) * 16;

    // stage A (64 rows x 64 bf16): 512 x 16B chunks
#pragma unroll
    for (int it = 0; it < 2; ++it) {
        const int c = it * 256 + tid;
        const int r = c >> 3, kc = c & 7;
        async_copy_b128(&As[r][kc * 8], zbf + (size_t)(blockM + r) * 64 + kc * 8);
    }
    auto stageB = [&](int ctile, int buf) {
#pragma unroll
        for (int it = 0; it < 8; ++it) {
            const int c = it * 256 + tid;
            const int n = c >> 3, kc = c & 7;
            async_copy_b128(&Bs[buf][n][kc * 8],
                            cbbf + (size_t)(ctile * CT + n) * 64 + kc * 8);
        }
    };
    stageB(0, 0);

    v16bf afrag[2][2];                              // [i][kstep], loaded once
    float bestv[2][8];
    int   besti[2][8];
#pragma unroll
    for (int i = 0; i < 2; ++i)
#pragma unroll
        for (int r = 0; r < 8; ++r) { bestv[i][r] = 3.4e38f; besti[i][r] = 0; }

    for (int ct = 0; ct < NT; ++ct) {
        const int cur = ct & 1;
        if (ct + 1 < NT) {
            stageB(ct + 1, cur ^ 1);
            asm volatile("s_wait_asynccnt %0" :: "n"(8) : "memory");
        } else {
            asm volatile("s_wait_asynccnt 0" ::: "memory");
        }
        __syncthreads();

        if (ct == 0) {                              // A landed with first wait
#pragma unroll
            for (int i = 0; i < 2; ++i)
#pragma unroll
                for (int ks = 0; ks < 2; ++ks) {
                    v8bf lo = *(const v8bf*)&As[rowBase + i * 16 + mA][ks * 32 + kHi];
                    v8bf hi = *(const v8bf*)&As[rowBase + i * 16 + mA][ks * 32 + 16 + kHi];
                    afrag[i][ks] = __builtin_shufflevector(lo, hi,
                        0,1,2,3,4,5,6,7,8,9,10,11,12,13,14,15);
                }
        }

        v8f acc[2][4] = {};
#pragma unroll
        for (int ks = 0; ks < 2; ++ks) {
            v16bf b[4];
#pragma unroll
            for (int j = 0; j < 4; ++j) {
                v8bf lo = *(const v8bf*)&Bs[cur][colBase + j * 16 + nB][ks * 32 + kHi];
                v8bf hi = *(const v8bf*)&Bs[cur][colBase + j * 16 + nB][ks * 32 + 16 + kHi];
                b[j] = __builtin_shufflevector(lo, hi,
                    0,1,2,3,4,5,6,7,8,9,10,11,12,13,14,15);
            }
#pragma unroll
            for (int i = 0; i < 2; ++i)
#pragma unroll
                for (int j = 0; j < 4; ++j)
                    acc[i][j] = __builtin_amdgcn_wmma_f32_16x16x32_bf16(
                        false, afrag[i][ks], false, b[j], (short)0, acc[i][j], false, false);
        }

        // score + running argmin
#pragma unroll
        for (int j = 0; j < 4; ++j) {
            const int colg = ct * CT + colBase + j * 16 + nB;
            const float eq = esq[colg];
#pragma unroll
            for (int i = 0; i < 2; ++i)
#pragma unroll
                for (int r = 0; r < 8; ++r) {
                    const float s = fmaf(-2.f, acc[i][j][r], eq);
                    if (s < bestv[i][r]) { bestv[i][r] = s; besti[i][r] = colg; }
                }
        }
        __syncthreads();
    }

    // reduce across the 16 lanes of each half-wave (prefer lower index on tie)
#pragma unroll
    for (int mask = 1; mask <= 8; mask <<= 1) {
#pragma unroll
        for (int i = 0; i < 2; ++i)
#pragma unroll
            for (int r = 0; r < 8; ++r) {
                const float ov = __shfl_xor(bestv[i][r], mask, 32);
                const int   oi = __shfl_xor(besti[i][r], mask, 32);
                if (ov < bestv[i][r] || (ov == bestv[i][r] && oi < besti[i][r])) {
                    bestv[i][r] = ov; besti[i][r] = oi;
                }
            }
    }
    if ((lane & 15) == 0) {
        const int mOff = (lane >> 4) * 8;
#pragma unroll
        for (int i = 0; i < 2; ++i)
#pragma unroll
            for (int r = 0; r < 8; ++r)
                besti_out[blockM + rowBase + i * 16 + mOff + r] = besti[i][r];
    }
}

// ---------------------------------------------------------------------------
// VQ finalize: gather z_q, write idx (float) and (1+beta)*MSE loss.
// ---------------------------------------------------------------------------
__global__ __launch_bounds__(256) void vq_finalize(
    const float* __restrict__ z_e, const float* __restrict__ codebook,
    const int* __restrict__ besti,
    float* __restrict__ z_q, unsigned short* __restrict__ zq_bf,
    float* __restrict__ idx_out, float* __restrict__ loss_acc)
{
    const int row = blockIdx.x * 256 + threadIdx.x;
    const int k = besti[row];
    idx_out[row] = (float)k;
    float lsum = 0.f;
#pragma unroll
    for (int i = 0; i < LATENT_DIM; i += 4) {
        const float4 q = *(const float4*)(codebook + (size_t)k * LATENT_DIM + i);
        const float4 z = *(const float4*)(z_e + (size_t)row * LATENT_DIM + i);
        *(float4*)(z_q + (size_t)row * LATENT_DIM + i) = q;
        uint2 p;
        p.x = (unsigned)f32_to_bf16_raw(q.x) | ((unsigned)f32_to_bf16_raw(q.y) << 16);
        p.y = (unsigned)f32_to_bf16_raw(q.z) | ((unsigned)f32_to_bf16_raw(q.w) << 16);
        *(uint2*)(zq_bf + (size_t)row * LATENT_DIM + i) = p;
        float d0 = q.x - z.x, d1 = q.y - z.y, d2 = q.z - z.z, d3 = q.w - z.w;
        lsum += fmaf(d0, d0, fmaf(d1, d1, fmaf(d2, d2, d3 * d3)));
    }
    lsum *= 1.25f / ((float)BATCH * (float)LATENT_DIM);
#pragma unroll
    for (int off = 16; off > 0; off >>= 1) lsum += __shfl_xor(lsum, off, 32);
    if ((threadIdx.x & 31) == 0) atomicAdd(loss_acc, lsum);
}

__global__ void zero_loss_kernel(float* p) { *p = 0.f; }

extern "C" void kernel_launch(void* const* d_in, const int* in_sizes, int n_in,
                              void* d_out, int out_size, void* d_ws, size_t ws_size,
                              hipStream_t stream) {
    const float* x        = (const float*)d_in[0];
    const float* enc_w1   = (const float*)d_in[1];
    const float* enc_b1   = (const float*)d_in[2];
    const float* enc_w2   = (const float*)d_in[3];
    const float* enc_b2   = (const float*)d_in[4];
    const float* codebook = (const float*)d_in[5];
    const float* dec_w1   = (const float*)d_in[6];
    const float* dec_b1   = (const float*)d_in[7];
    const float* dec_w2   = (const float*)d_in[8];
    const float* dec_b2   = (const float*)d_in[9];

    float* out     = (float*)d_out;
    float* x_recon = out;                                   // [B, 1024]
    float* z_e     = x_recon + (size_t)BATCH * INPUT_DIM;   // [B, 64]
    float* z_q     = z_e     + (size_t)BATCH * LATENT_DIM;  // [B, 64]
    float* idxf    = z_q     + (size_t)BATCH * LATENT_DIM;  // [B]
    float* loss    = idxf    + BATCH;                       // [1]

    // Workspace (raw bf16 unless noted):
    unsigned short* hbuf = (unsigned short*)d_ws;                        // [B,2048] h/hd
    unsigned short* xbf  = hbuf + (size_t)BATCH * HIDDEN_DIM;            // [B,1024]
    unsigned short* zebf = xbf  + (size_t)BATCH * INPUT_DIM;             // [B,64]
    unsigned short* zqbf = zebf + (size_t)BATCH * LATENT_DIM;            // [B,64]
    unsigned short* w1t  = zqbf + (size_t)BATCH * LATENT_DIM;            // [2048,1024]
    unsigned short* w2t  = w1t  + (size_t)INPUT_DIM * HIDDEN_DIM;        // [64,2048]
    unsigned short* d1t  = w2t  + (size_t)HIDDEN_DIM * LATENT_DIM;       // [2048,64]
    unsigned short* d2t  = d1t  + (size_t)LATENT_DIM * HIDDEN_DIM;       // [1024,2048]
    unsigned short* cbbf = d2t  + (size_t)HIDDEN_DIM * INPUT_DIM;        // [4096,64]
    float*          esq  = (float*)(cbbf + (size_t)NUM_EMB * LATENT_DIM);// [4096] f32
    int*            bidx = (int*)(esq + NUM_EMB);                        // [B] i32

    // 0) pre-convert: x, codebook -> bf16; weights -> bf16 [N,K]; esq
    f32_to_bf16_kernel<<<(BATCH * INPUT_DIM) / 2048, 256, 0, stream>>>(x, xbf);
    f32_to_bf16_kernel<<<(NUM_EMB * LATENT_DIM) / 2048, 256, 0, stream>>>(codebook, cbbf);
    codebook_esq_kernel<<<NUM_EMB / 256, 256, 0, stream>>>(codebook, esq);
    transpose_to_bf16<<<dim3(HIDDEN_DIM / 32, INPUT_DIM / 32),  256, 0, stream>>>(enc_w1, w1t, INPUT_DIM,  HIDDEN_DIM);
    transpose_to_bf16<<<dim3(LATENT_DIM / 32, HIDDEN_DIM / 32), 256, 0, stream>>>(enc_w2, w2t, HIDDEN_DIM, LATENT_DIM);
    transpose_to_bf16<<<dim3(HIDDEN_DIM / 32, LATENT_DIM / 32), 256, 0, stream>>>(dec_w1, d1t, LATENT_DIM, HIDDEN_DIM);
    transpose_to_bf16<<<dim3(INPUT_DIM / 32, HIDDEN_DIM / 32),  256, 0, stream>>>(dec_w2, d2t, HIDDEN_DIM, INPUT_DIM);

    // 1) h = relu(x @ enc_w1 + enc_b1)  -> bf16 ws
    gemm_wmma<64, 256, 2, 4, true, true>
        <<<dim3(HIDDEN_DIM / 256, BATCH / 64), 256, 0, stream>>>(
            xbf, w1t, enc_b1, (void*)hbuf, BATCH, HIDDEN_DIM, INPUT_DIM);

    // 2) z_e = h @ enc_w2 + enc_b2  -> f32 d_out, then bf16 copy
    gemm_wmma<128, 64, 2, 2, false, false>
        <<<dim3(LATENT_DIM / 64, BATCH / 128), 256, 0, stream>>>(
            hbuf, w2t, enc_b2, (void*)z_e, BATCH, LATENT_DIM, HIDDEN_DIM);
    f32_to_bf16_kernel<<<(BATCH * LATENT_DIM) / 2048, 256, 0, stream>>>(z_e, zebf);

    // 3) VQ: WMMA scoring + argmin, then gather/loss
    zero_loss_kernel<<<1, 1, 0, stream>>>(loss);
    vq_score_argmin<<<BATCH / 64, 256, 0, stream>>>(zebf, cbbf, esq, bidx);
    vq_finalize<<<BATCH / 256, 256, 0, stream>>>(z_e, codebook, bidx, z_q, zqbf, idxf, loss);

    // 4) hd = relu(z_q @ dec_w1 + dec_b1)  -> bf16 ws (reuse hbuf)
    gemm_wmma<64, 256, 2, 4, true, true>
        <<<dim3(HIDDEN_DIM / 256, BATCH / 64), 256, 0, stream>>>(
            zqbf, d1t, dec_b1, (void*)hbuf, BATCH, HIDDEN_DIM, LATENT_DIM);

    // 5) x_recon = hd @ dec_w2 + dec_b2  -> f32 d_out
    gemm_wmma<64, 256, 2, 4, false, false>
        <<<dim3(INPUT_DIM / 256, BATCH / 64), 256, 0, stream>>>(
            hbuf, d2t, dec_b2, (void*)x_recon, BATCH, INPUT_DIM, HIDDEN_DIM);
}
